// rn_layer_75445395521744
// MI455X (gfx1250) — matrix operand order repaired and checked by reading, not compiled, not run on hardware
//
#include <hip/hip_runtime.h>
#include <hip/hip_bf16.h>

// ---------------------------------------------------------------------------
// Relation Network for MI455X (gfx1250, wave32, WMMA).
//   B=32, NUM=64, LENTH=64, AUX=64, IN_G=194, SIZE=OUT=256.
// g1 is decomposed over the broadcast/concat structure of `pairs`, so the
// heavy work is 3x [131072 x 256] @ [256 x 256] GEMMs on v_wmma_f32_16x16x32_f16.
// Layers are instantiated separately (no pointer select) so weight streams
// lower to global_load_b128 instead of flat_load_b128.
// ---------------------------------------------------------------------------

typedef _Float16 half16 __attribute__((ext_vector_type(16)));
typedef _Float16 half8  __attribute__((ext_vector_type(8)));
typedef float    float8 __attribute__((ext_vector_type(8)));

#define HB_STRIDE 272   // 256 + 16 halves; keeps rows 16B-aligned (544B)

// --------------------------- zero accumulator ------------------------------
__global__ __launch_bounds__(256) void rn_zero(float* __restrict__ p, int n) {
    int i = blockIdx.x * 256 + threadIdx.x;
    if (i < n) p[i] = 0.f;
}

// --------------------------- g1 decomposition ------------------------------
// rowA[bn][c] = x[b,n] (64 dims + coord=n) @ g1_w[0:65]
// rowC[bn][c] = xa[b,n] (128 dims + coord=n) @ g1_w[65:194] + g1_b
__global__ __launch_bounds__(256) void rn_prep(
    const float* __restrict__ x_aux, const float* __restrict__ g1w,
    const float* __restrict__ g1b,
    float* __restrict__ rowA, float* __restrict__ rowC)
{
    __shared__ float xr[128];
    const int bn = blockIdx.x;     // b*64 + n
    const int n  = bn & 63;
    const int c  = threadIdx.x;    // output column 0..255
    if (c < 128) xr[c] = x_aux[(size_t)bn * 128 + c];
    __syncthreads();

    float a = (float)n * g1w[64 * 256 + c];          // coord row of x-part
    #pragma unroll 4
    for (int k = 0; k < 64; ++k) a += xr[k] * g1w[k * 256 + c];

    float cc = g1b[c] + (float)n * g1w[193 * 256 + c]; // coord row of xa-part
    #pragma unroll 4
    for (int k = 0; k < 128; ++k) cc += xr[k] * g1w[(65 + k) * 256 + c];

    rowA[(size_t)bn * 256 + c] = a;
    rowC[(size_t)bn * 256 + c] = cc;
}

// --------------------------- weight pre-swizzle ----------------------------
// Packed B-fragment layout for V_WMMA_F32_16X16X32_F16:
//   frag(kb, nb, lane)[e] = W[kb*32 + (lane<16?0:16) + e][nb*16 + (lane&15)]
// stored contiguously: [((nb*8 + kb)*32 + lane)*16 + e]
__global__ __launch_bounds__(256) void rn_pack(
    const float* __restrict__ w2, const float* __restrict__ w3,
    const float* __restrict__ w4,
    _Float16* __restrict__ p2, _Float16* __restrict__ p3,
    _Float16* __restrict__ p4)
{
    const int flat = blockIdx.x * 256 + threadIdx.x;   // 0..12287
    const int mat  = flat >> 12;
    const int rem  = flat & 4095;
    const int nb   = rem >> 8;          // 0..15
    const int kb   = (rem >> 5) & 7;    // 0..7
    const int lane = rem & 31;
    const float*    W = (mat == 0) ? w2 : (mat == 1) ? w3 : w4;
    _Float16*       P = (mat == 0) ? p2 : (mat == 1) ? p3 : p4;
    const int col   = nb * 16 + (lane & 15);
    const int kbase = kb * 32 + ((lane >> 4) << 4);
    _Float16* dst = P + (((size_t)(nb * 8 + kb) * 32 + lane) << 4);
    #pragma unroll
    for (int e = 0; e < 16; ++e)
        dst[e] = (_Float16)W[(size_t)(kbase + e) * 256 + col];
}

// --------------------------- one fused g-layer -----------------------------
// 64x256 tile in LDS (f16) x packed 256x256 weights -> bias+relu.
// LAST=false: write result back to hbuf (f16).
// LAST=true : relu-sum over the tile's 64 rows into red[256] (ds_add_f32).
template <bool LAST>
__device__ __forceinline__ void g_layer(
    const _Float16* __restrict__ W, const float* __restrict__ bias,
    _Float16* hbuf, float* red,
    int lane, int m, int hi, int rb, int ch, int arow, int aklo)
{
    float8 acc[8];
    #pragma unroll
    for (int n2 = 0; n2 < 8; ++n2)
        #pragma unroll
        for (int v = 0; v < 8; ++v) acc[n2][v] = 0.f;

    #pragma unroll
    for (int kb = 0; kb < 8; ++kb) {
        // A fragment: two 16B ds_read_b128 per lane
        const _Float16* ap = &hbuf[arow * HB_STRIDE + kb * 32 + aklo];
        half8 alo = *(const half8*)(ap);
        half8 ahi = *(const half8*)(ap + 8);
        half16 afrag;
        #pragma unroll
        for (int e = 0; e < 8; ++e) { afrag[e] = alo[e]; afrag[e + 8] = ahi[e]; }

        #pragma unroll
        for (int n2 = 0; n2 < 8; ++n2) {
            const int nb = ch * 8 + n2;
            half16 bfrag = *(const half16*)(W + (((size_t)(nb * 8 + kb) * 32 + lane) << 4));
            acc[n2] = __builtin_amdgcn_wmma_f32_16x16x32_f16(
                false, afrag, false, bfrag, (short)0, acc[n2], false, false);
        }
    }
    __syncthreads();   // all waves finished reading hbuf

    if (!LAST) {
        // bias + relu -> f16 LDS (in place)
        #pragma unroll
        for (int n2 = 0; n2 < 8; ++n2) {
            const int col = ch * 128 + n2 * 16 + m;
            const float bv = bias[col];
            #pragma unroll
            for (int v = 0; v < 8; ++v) {
                const int row = rb * 16 + hi * 8 + v;
                hbuf[row * HB_STRIDE + col] =
                    (_Float16)fmaxf(acc[n2][v] + bv, 0.f);
            }
        }
    } else {
        // g4: bias + relu + sum over the tile's 64 pair-rows
        #pragma unroll
        for (int n2 = 0; n2 < 8; ++n2) {
            const int col = ch * 128 + n2 * 16 + m;
            const float bv = bias[col];
            float s = 0.f;
            #pragma unroll
            for (int v = 0; v < 8; ++v) s += fmaxf(acc[n2][v] + bv, 0.f);
            atomicAdd(&red[col], s);   // ds_add_f32
        }
    }
    __syncthreads();
}

// --------------------------- main fused g-MLP ------------------------------
// One workgroup = one (b, i=t) tile of 64 pair-rows (j = 0..63) x 256 cols.
// 8 waves: wave w -> row-block rb = w&3 (16 rows), col-half ch = w>>2 (128 cols).
__global__ __launch_bounds__(256) void rn_g_main(
    const float* __restrict__ rowA, const float* __restrict__ rowC,
    const _Float16* __restrict__ pw2, const _Float16* __restrict__ pw3,
    const _Float16* __restrict__ pw4,
    const float* __restrict__ b2, const float* __restrict__ b3,
    const float* __restrict__ b4,
    float* __restrict__ xg)
{
    __shared__ __align__(16) _Float16 hbuf[64 * HB_STRIDE];
    __shared__ float red[256];

    const int tid  = threadIdx.x;
    const int wave = tid >> 5;
    const int lane = tid & 31;
    const int b    = blockIdx.x >> 6;   // batch
    const int t    = blockIdx.x & 63;   // i index (constant over the tile)

    // h1 = relu(rowA[b, j] + rowC[b, t]) -> f16 LDS
    {
        const int r  = tid >> 2;            // j row 0..63
        const int c0 = (tid & 3) * 64;
        const float4* pa = (const float4*)(rowA + ((size_t)(b * 64 + r)) * 256 + c0);
        const float4* pc = (const float4*)(rowC + ((size_t)(b * 64 + t)) * 256 + c0);
        _Float16* hd = &hbuf[r * HB_STRIDE + c0];
        #pragma unroll
        for (int q = 0; q < 16; ++q) {
            float4 a = pa[q];
            float4 c = pc[q];
            hd[q * 4 + 0] = (_Float16)fmaxf(a.x + c.x, 0.f);
            hd[q * 4 + 1] = (_Float16)fmaxf(a.y + c.y, 0.f);
            hd[q * 4 + 2] = (_Float16)fmaxf(a.z + c.z, 0.f);
            hd[q * 4 + 3] = (_Float16)fmaxf(a.w + c.w, 0.f);
        }
        red[tid] = 0.f;
    }
    __syncthreads();

    const int rb   = wave & 3;
    const int ch   = wave >> 2;
    const int m    = lane & 15;
    const int hi   = lane >> 4;         // 0 or 1
    const int arow = rb * 16 + m;
    const int aklo = hi * 16;           // A-frag K sub-offset per lane half

    // Three g-layers, each with a statically-known global weight pointer.
    g_layer<false>(pw2, b2, hbuf, red, lane, m, hi, rb, ch, arow, aklo);
    g_layer<false>(pw3, b3, hbuf, red, lane, m, hi, rb, ch, arow, aklo);
    g_layer<true >(pw4, b4, hbuf, red, lane, m, hi, rb, ch, arow, aklo);

    atomicAdd(&xg[b * 256 + tid], red[tid]);   // global_atomic_add_f32
}

// --------------------------- f-MLP (tiny) ----------------------------------
__global__ __launch_bounds__(256) void rn_f(
    const float* __restrict__ xg,
    const float* __restrict__ f1w, const float* __restrict__ f1b,
    const float* __restrict__ f2w, const float* __restrict__ f2b,
    const float* __restrict__ f3w, const float* __restrict__ f3b,
    float* __restrict__ out)
{
    __shared__ float v0[256], v1[256];
    const int b = blockIdx.x, c = threadIdx.x;
    v0[c] = xg[b * 256 + c];
    __syncthreads();

    float s = f1b[c];
    #pragma unroll 4
    for (int k = 0; k < 256; ++k) s += v0[k] * f1w[k * 256 + c];
    v1[c] = fmaxf(s, 0.f);
    __syncthreads();

    s = f2b[c];
    #pragma unroll 4
    for (int k = 0; k < 256; ++k) s += v1[k] * f2w[k * 256 + c];
    v0[c] = fmaxf(s, 0.f);
    __syncthreads();

    s = f3b[c];
    #pragma unroll 4
    for (int k = 0; k < 256; ++k) s += v0[k] * f3w[k * 256 + c];
    out[b * 256 + c] = s;
}

// ---------------------------------------------------------------------------
extern "C" void kernel_launch(void* const* d_in, const int* in_sizes, int n_in,
                              void* d_out, int out_size, void* d_ws, size_t ws_size,
                              hipStream_t stream) {
    (void)in_sizes; (void)n_in; (void)out_size; (void)ws_size;
    const float* x_aux = (const float*)d_in[0];
    const float* g1w = (const float*)d_in[1];  const float* g1b = (const float*)d_in[2];
    const float* g2w = (const float*)d_in[3];  const float* g2b = (const float*)d_in[4];
    const float* g3w = (const float*)d_in[5];  const float* g3b = (const float*)d_in[6];
    const float* g4w = (const float*)d_in[7];  const float* g4b = (const float*)d_in[8];
    const float* f1w = (const float*)d_in[9];  const float* f1b = (const float*)d_in[10];
    const float* f2w = (const float*)d_in[11]; const float* f2b = (const float*)d_in[12];
    const float* f3w = (const float*)d_in[13]; const float* f3b = (const float*)d_in[14];

    char* ws = (char*)d_ws;
    float*    rowA = (float*)(ws);                       // 2048*256*4 = 2 MB
    float*    rowC = (float*)(ws + (2u << 20));          // 2 MB
    _Float16* p2   = (_Float16*)(ws + (4u << 20));       // 128 KB
    _Float16* p3   = (_Float16*)(ws + (4u << 20) + 131072u);
    _Float16* p4   = (_Float16*)(ws + (4u << 20) + 262144u);
    float*    xgac = (float*)(ws + (4u << 20) + 393216u); // 32*256*4 = 32 KB

    rn_zero<<<32, 256, 0, stream>>>(xgac, 32 * 256);
    rn_prep<<<2048, 256, 0, stream>>>(x_aux, g1w, g1b, rowA, rowC);
    rn_pack<<<48, 256, 0, stream>>>(g2w, g3w, g4w, p2, p3, p4);
    rn_g_main<<<2048, 256, 0, stream>>>(rowA, rowC, p2, p3, p4,
                                        g2b, g3b, g4b, xgac);
    rn_f<<<32, 256, 0, stream>>>(xgac, f1w, f1b, f2w, f2b, f3w, f3b,
                                 (float*)d_out);
}
